// SparseCNN_TensorIO_23656679866690
// MI455X (gfx1250) — matrix operand order repaired and checked by reading, not compiled, not run on hardware
//
#include <hip/hip_runtime.h>

typedef __attribute__((ext_vector_type(16))) _Float16 v16h;
typedef __attribute__((ext_vector_type(8)))  _Float16 v8h;
typedef __attribute__((ext_vector_type(8)))  float    v8f;

#define NB   8
#define MM   2048
#define GRID 125
#define NC   (GRID*GRID)
#define CH   64
#define ACH  128
#define DH   32

#define WMMA16(a,b,c) __builtin_amdgcn_wmma_f32_16x16x32_f16(false,(a),false,(b),(short)0,(c),false,false)

union frag_u { v16h v; v8h h[2]; };

// ---------- WMMA fragment loaders (vectorized, per CDNA5 ISA §7.12.2) ----------

// A fragment 16x32 f16, element (m,k) at base[m*ld + k].
// Lane layout: lanes 0-15 row m=lane, K {0..7,16..23}; lanes 16-31 row m=lane-16, K {8..15,24..31}.
// Each half is 8 contiguous halves -> two b128 loads. base must be 16B aligned, ld multiple of 8.
__device__ inline v16h load_a_frag(const _Float16* base, int ld, int lane) {
  int m = lane & 15, khi = lane >> 4;
  const _Float16* p = base + m * ld + khi * 8;
  frag_u u;
  u.h[0] = *(const v8h*)(p);
  u.h[1] = *(const v8h*)(p + 16);
  return u.v;
}
// B fragment 32x16 f16 from K-contiguous source: element (k,n) at base[n*ld + k].
// Lane layout: lanes 0-15 col n=lane, K {0..7,16..23}; lanes 16-31 col n=lane-16, K {8..15,24..31}.
// Each half is 8 contiguous halves -> two b128 loads.
__device__ inline v16h load_bK_frag(const _Float16* base, int ld, int lane) {
  int nn = lane & 15, khi = lane >> 4;
  const _Float16* p = base + nn * ld + khi * 8;
  frag_u u;
  u.h[0] = *(const v8h*)(p);
  u.h[1] = *(const v8h*)(p + 16);
  return u.v;
}

// ---------- Stage 1: ordered compaction of active bottleneck cells ----------
__global__ void compact_kernel(const float* __restrict__ x, int* __restrict__ idx,
                               int* __restrict__ n_active) {
  int b = blockIdx.x;
  const float* xb = x + (size_t)b * 500 * 500;
  __shared__ int scan[256];
  __shared__ int base;
  if (threadIdx.x == 0) base = 0;
  __syncthreads();
  for (int s0 = 0; s0 < NC; s0 += 256) {
    int cell = s0 + threadIdx.x;
    int f = 0;
    if (cell < NC) {
      int r = cell / GRID, c = cell % GRID;
      for (int dy = 0; dy < 4 && !f; ++dy)
        for (int dx = 0; dx < 4; ++dx)
          if (xb[(4 * r + dy) * 500 + 4 * c + dx] != 0.0f) { f = 1; break; }
    }
    scan[threadIdx.x] = f;
    __syncthreads();
    for (int off = 1; off < 256; off <<= 1) {
      int v = (threadIdx.x >= off) ? scan[threadIdx.x - off] : 0;
      __syncthreads();
      scan[threadIdx.x] += v;
      __syncthreads();
    }
    int slot = base + scan[threadIdx.x] - f;
    if (f && slot < MM) idx[b * MM + slot] = cell;
    __syncthreads();
    if (threadIdx.x == 255) base += scan[255];
    __syncthreads();
  }
  int n = base < MM ? base : MM;
  if (threadIdx.x == 0) n_active[b] = n;
  for (int j = n + threadIdx.x; j < MM; j += 256) idx[b * MM + j] = -1;
}

// ---------- Stage 2: sparse conv chain per active token (64 thr / token) ----------
__global__ void feats_kernel(const float* __restrict__ x,
                             const float* w0, const float* b0, const float* g0, const float* be0,
                             const float* w1, const float* b1, const float* g1, const float* be1,
                             const float* w2, const float* b2, const float* g2, const float* be2,
                             const int* __restrict__ idx, const int* __restrict__ n_active,
                             float* __restrict__ feats, float* __restrict__ pos) {
  int b = blockIdx.x >> 11;
  int j = blockIdx.x & (MM - 1);
  int t = threadIdx.x;
  float* fout = feats + ((size_t)(b * MM + j)) * CH;
  float* pout = pos + (size_t)(b * MM + j) * 2;
  int n = n_active[b];
  if (j >= n) { fout[t] = 0.0f; if (t < 2) pout[t] = 0.0f; return; }
  int cell = idx[b * MM + j];
  int r = cell / GRID, c = cell % GRID;
  __shared__ float xs[6][6];
  __shared__ float y0[16][32];
  __shared__ float y1[4][64];
  if (t < 36) {
    int dy = t / 6, dx = t % 6;
    int iy = 4 * r - 1 + dy, ix = 4 * c - 1 + dx;
    float v = 0.0f;
    if (iy >= 0 && iy < 500 && ix >= 0 && ix < 500) v = x[((size_t)b * 500 + iy) * 500 + ix];
    xs[dy][dx] = v;
  }
  __syncthreads();
  const float bns = rsqrtf(1.0f + 1e-5f);
  { // y0: 16 positions x 32 ch (submanifold 3x3)
    int ch = t & 31;
    int ph = (t >> 5) * 8;
    for (int pp = 0; pp < 8; ++pp) {
      int p = ph + pp, pr = p >> 2, pc = p & 3;
      float v = 0.0f;
      if (xs[pr + 1][pc + 1] != 0.0f) {
        float acc = b0[ch];
        for (int ky = 0; ky < 3; ++ky)
          for (int kx = 0; kx < 3; ++kx)
            acc += xs[pr + ky][pc + kx] * w0[(ky * 3 + kx) * 32 + ch];
        v = g0[ch] * acc * bns + be0[ch];
        v = v > 0.0f ? v : 0.0f;
      }
      y0[p][ch] = v;
    }
  }
  __syncthreads();
  { // y1: 4 positions x 64 ch (2x2 stride-2)
    int ch = t;
    for (int q = 0; q < 4; ++q) {
      int qr = q >> 1, qc = q & 1;
      bool act = (xs[2*qr+1][2*qc+1] != 0.f) || (xs[2*qr+1][2*qc+2] != 0.f) ||
                 (xs[2*qr+2][2*qc+1] != 0.f) || (xs[2*qr+2][2*qc+2] != 0.f);
      float v = 0.0f;
      if (act) {
        float acc = b1[ch];
        for (int dy = 0; dy < 2; ++dy)
          for (int dx = 0; dx < 2; ++dx) {
            int p = (2 * qr + dy) * 4 + (2 * qc + dx);
            const float* wrow = w1 + ((dy * 2 + dx) * 32) * 64 + ch;
            for (int ci = 0; ci < 32; ++ci) acc += y0[p][ci] * wrow[ci * 64];
          }
        v = g1[ch] * acc * bns + be1[ch];
        v = v > 0.0f ? v : 0.0f;
      }
      y1[q][ch] = v;
    }
  }
  __syncthreads();
  { // y2: 64 ch, known-active cell
    int ch = t;
    float acc = b2[ch];
    for (int q = 0; q < 4; ++q) {
      const float* wrow = w2 + (q * 64) * 64 + ch;
      for (int ci = 0; ci < 64; ++ci) acc += y1[q][ci] * wrow[ci * 64];
    }
    float v = g2[ch] * acc * bns + be2[ch];
    fout[ch] = v > 0.0f ? v : 0.0f;
  }
  if (t == 0) pout[0] = (float)r / 125.0f;
  if (t == 1) pout[1] = (float)c / 125.0f;
}

// ---------- weights f32 -> f16, stored transposed [N][K] so B-frags are K-contiguous ----------
__global__ void wconv_kernel(const float* wq, const float* wk, const float* wv, const float* wo,
                             _Float16* wqT, _Float16* wkT, _Float16* wvT, _Float16* woT) {
  int tid = blockIdx.x * 256 + threadIdx.x;   // 4 * 16384
  int seg = tid >> 14, off = tid & 16383;
  int l = off >> 13, rem = off & 8191;
  if (seg < 3) {
    // (2,64,128): src[l][k][n] -> dst[l][n*64 + k]
    int k = rem >> 7, nn = rem & 127;
    const float* s = (seg == 0) ? wq : (seg == 1) ? wk : wv;
    _Float16* d = (seg == 0) ? wqT : (seg == 1) ? wkT : wvT;
    d[l * 8192 + nn * 64 + k] = (_Float16)s[off];
  } else {
    // (2,128,64): src[l][k][n] -> dst[l][n*128 + k]
    int k = rem >> 6, nn = rem & 63;
    woT[l * 8192 + nn * 128 + k] = (_Float16)wo[off];
  }
}

// ---------- h = feats + pos@wpe + bpe (f16) ----------
__global__ void h_kernel(const float* __restrict__ feats, const float* __restrict__ pos,
                         const int* __restrict__ n_active, const float* wpe, const float* bpe,
                         int layer, _Float16* __restrict__ h16) {
  int tid = blockIdx.x * 256 + threadIdx.x;   // NB*MM*CH
  int ch = tid & 63;
  int bj = tid >> 6;
  int b = bj >> 11;
  int j = bj & (MM - 1);
  float v = 0.0f;
  if (j < n_active[b]) {
    float p0 = pos[(size_t)bj * 2 + 0], p1 = pos[(size_t)bj * 2 + 1];
    v = feats[tid] + p0 * wpe[(layer * 2 + 0) * 64 + ch] + p1 * wpe[(layer * 2 + 1) * 64 + ch]
        + bpe[layer * 64 + ch];
  }
  h16[tid] = (_Float16)v;
}

// ---------- q/k/v projections: WMMA GEMM [M,64]x[64,128] ----------
// q,k stored row-major [M,128]; v stored transposed vT[b][head][dh][M] (K-contiguous for P@V).
__global__ void qkv_gemm(const _Float16* __restrict__ h16,
                         const _Float16* wqT, const _Float16* wkT, const _Float16* wvT,
                         const float* bq, const float* bk, const float* bv, int layer,
                         _Float16* q16, _Float16* k16, _Float16* vT) {
  int wave = (blockIdx.x * blockDim.x + threadIdx.x) >> 5;  // 24576 waves
  int lane = threadIdx.x & 31;
  int ct = wave & 7;
  int rt = (wave >> 3) & 127;
  int mat = (wave >> 10) % 3;
  int b = wave / 3072;
  const _Float16* w; const float* bias;
  if (mat == 0) { w = wqT; bias = bq; }
  else if (mat == 1) { w = wkT; bias = bk; }
  else { w = wvT; bias = bv; }
  const _Float16* A = h16 + ((size_t)b * MM + rt * 16) * CH;
  const _Float16* Bw = w + layer * 8192 + ct * 16 * CH;   // [N][K=64]
  v8f acc = {};
#pragma unroll
  for (int ks = 0; ks < 2; ++ks) {
    v16h a = load_a_frag(A + ks * 32, CH, lane);
    v16h bf = load_bK_frag(Bw + ks * 32, CH, lane);
    acc = WMMA16(a, bf, acc);
  }
  int col = ct * 16 + (lane & 15);
  int khi = lane >> 4;
  float bb = bias[layer * ACH + col];
  if (mat == 2) {
    v8h pk;
#pragma unroll
    for (int i = 0; i < 8; ++i) pk[i] = (_Float16)(acc[i] + bb);
    // vT[((b*4 + head)*32 + d)*M + row], rows rt*16+khi*8 .. +7 contiguous -> one b128 store
    _Float16* vt = vT + (((size_t)b * 4 + (col >> 5)) * 32 + (col & 31)) * MM + rt * 16 + khi * 8;
    *(v8h*)vt = pk;
  } else {
    _Float16* out = ((mat == 0) ? q16 : k16) + (size_t)b * MM * ACH;
#pragma unroll
    for (int i = 0; i < 8; ++i) {
      int row = rt * 16 + i + 8 * khi;
      out[(size_t)row * ACH + col] = (_Float16)(acc[i] + bb);
    }
  }
}

// ---------- flash attention: one wave per (b, head, 16-query tile) ----------
__global__ void flash_kernel(const _Float16* __restrict__ q16, const _Float16* __restrict__ k16,
                             const _Float16* __restrict__ vT, const int* __restrict__ n_active,
                             _Float16* __restrict__ o16) {
  int lane = threadIdx.x;
  int bid = blockIdx.x;            // b*512 + h*128 + qt
  int qt = bid & 127;
  int h = (bid >> 7) & 3;
  int b = bid >> 9;
  int n = n_active[b];
  if (qt * 16 >= n) return;
  const _Float16* qb = q16 + ((size_t)b * MM + qt * 16) * ACH + h * DH;
  const _Float16* kb = k16 + (size_t)b * MM * ACH + h * DH;
  const _Float16* vb = vT + (((size_t)b * 4 + h) * 32) * MM;   // [dh][M]
  v16h qa = load_a_frag(qb, ACH, lane);
  v8f O0 = {}, O1 = {};
  float ms[8], ls[8];
#pragma unroll
  for (int i = 0; i < 8; ++i) { ms[i] = -1e30f; ls[i] = 0.0f; }
  __shared__ alignas(16) _Float16 Pl[16 * 32];
  const float scale = 0.17677669529663687f;   // 1/sqrt(32)
  int colg = lane & 15, khi = lane >> 4;
  int iters = (n + 31) >> 5;
  for (int it = 0; it < iters; ++it) {
    int k0 = it * 32;
    if (it + 1 < iters) {   // prefetch next K tile (one row per lane) and next V columns
      __builtin_prefetch(kb + (size_t)(k0 + 32 + lane) * ACH, 0, 3);
      __builtin_prefetch(vb + (size_t)lane * MM + k0 + 32, 0, 3);
    }
    // K^T fragment: element (k,n) = k16[key=k0+n][dim k] -> K-contiguous in row-major k16
    v16h kf0 = load_bK_frag(kb + (size_t)k0 * ACH, ACH, lane);
    v16h kf1 = load_bK_frag(kb + (size_t)(k0 + 16) * ACH, ACH, lane);
    v8f S0 = {}, S1 = {};
    S0 = WMMA16(qa, kf0, S0);
    S1 = WMMA16(qa, kf1, S1);
#pragma unroll
    for (int i = 0; i < 8; ++i) {
      float s0 = S0[i] * scale; if (k0 + colg >= n)      s0 = -1e9f;
      float s1 = S1[i] * scale; if (k0 + 16 + colg >= n) s1 = -1e9f;
      S0[i] = s0; S1[i] = s1;
    }
#pragma unroll
    for (int i = 0; i < 8; ++i) {
      float rm = fmaxf(S0[i], S1[i]);
      for (int off = 1; off < 16; off <<= 1) rm = fmaxf(rm, __shfl_xor(rm, off, 32));
      float mnew = fmaxf(ms[i], rm);
      float alpha = __expf(ms[i] - mnew);
      float p0 = __expf(S0[i] - mnew);
      float p1 = __expf(S1[i] - mnew);
      float rs = p0 + p1;
      for (int off = 1; off < 16; off <<= 1) rs += __shfl_xor(rs, off, 32);
      ls[i] = ls[i] * alpha + rs;
      ms[i] = mnew;
      O0[i] *= alpha; O1[i] *= alpha;
      S0[i] = p0; S1[i] = p1;
    }
#pragma unroll
    for (int i = 0; i < 8; ++i) {
      Pl[(i + 8 * khi) * 32 + colg]      = (_Float16)S0[i];
      Pl[(i + 8 * khi) * 32 + colg + 16] = (_Float16)S1[i];
    }
    __syncthreads();
    v16h pa = load_a_frag(Pl, 32, lane);
    __syncthreads();
    // V fragment: element (k,n) = vT[dh=n][key=k0+k] -> K-contiguous
    v16h vf0 = load_bK_frag(vb + k0, MM, lane);
    v16h vf1 = load_bK_frag(vb + (size_t)16 * MM + k0, MM, lane);
    O0 = WMMA16(pa, vf0, O0);
    O1 = WMMA16(pa, vf1, O1);
  }
  _Float16* ob = o16 + ((size_t)b * MM + qt * 16) * ACH + h * DH;
#pragma unroll
  for (int i = 0; i < 8; ++i) {
    float inv = 1.0f / ls[i];
    int row = i + 8 * khi;
    ob[(size_t)row * ACH + colg]      = (_Float16)(O0[i] * inv);
    ob[(size_t)row * ACH + colg + 16] = (_Float16)(O1[i] * inv);
  }
}

// ---------- output projection + residual + valid mask ----------
__global__ void oproj_gemm(const _Float16* __restrict__ o16, const _Float16* __restrict__ woT,
                           const float* bo, const int* __restrict__ n_active, int layer,
                           float* __restrict__ feats) {
  int wave = (blockIdx.x * blockDim.x + threadIdx.x) >> 5;  // 4096 waves
  int lane = threadIdx.x & 31;
  int ct = wave & 3;
  int rt = (wave >> 2) & 127;
  int b = wave >> 9;
  int n = n_active[b];
  if (rt * 16 >= n) return;
  const _Float16* A = o16 + ((size_t)b * MM + rt * 16) * ACH;
  const _Float16* Bw = woT + layer * 8192 + ct * 16 * ACH;   // [N][K=128]
  v8f acc = {};
#pragma unroll
  for (int ks = 0; ks < 4; ++ks) {
    v16h a = load_a_frag(A + ks * 32, ACH, lane);
    v16h bf = load_bK_frag(Bw + ks * 32, ACH, lane);
    acc = WMMA16(a, bf, acc);
  }
  int col = ct * 16 + (lane & 15);
  int khi = lane >> 4;
  float bb = bo[layer * CH + col];
  float* F = feats + ((size_t)b * MM + rt * 16) * CH;
#pragma unroll
  for (int i = 0; i < 8; ++i) {
    int row = i + 8 * khi;
    float valid = ((rt * 16 + row) < n) ? 1.0f : 0.0f;
    F[(size_t)row * CH + col] = (F[(size_t)row * CH + col] + acc[i] + bb) * valid;
  }
}

// ---------- pooled head ----------
__global__ void pool_head(const float* __restrict__ feats, const int* __restrict__ n_active,
                          const float* wh, const float* bh, float* __restrict__ out) {
  int b = blockIdx.x, t = threadIdx.x;   // 64 threads
  int n = n_active[b];
  float s = 0.0f;
  for (int j = 0; j < n; ++j) s += feats[((size_t)b * MM + j) * CH + t];
  __shared__ float ls[64];
  ls[t] = s;
  __syncthreads();
  if (t < 4) {
    float cnt = (float)(n > 0 ? n : 1);
    float acc = bh[t];
    for (int c = 0; c < 64; ++c) acc += (ls[c] / cnt) * wh[c * 4 + t];
    out[b * 4 + t] = acc;
  }
}

extern "C" void kernel_launch(void* const* d_in, const int* in_sizes, int n_in,
                              void* d_out, int out_size, void* d_ws, size_t ws_size,
                              hipStream_t stream) {
  (void)in_sizes; (void)n_in; (void)out_size; (void)ws_size;
  const float* x   = (const float*)d_in[0];
  const float* w0  = (const float*)d_in[1];
  const float* b0  = (const float*)d_in[2];
  const float* g0  = (const float*)d_in[3];
  const float* be0 = (const float*)d_in[4];
  const float* w1  = (const float*)d_in[5];
  const float* b1  = (const float*)d_in[6];
  const float* g1  = (const float*)d_in[7];
  const float* be1 = (const float*)d_in[8];
  const float* w2  = (const float*)d_in[9];
  const float* b2  = (const float*)d_in[10];
  const float* g2  = (const float*)d_in[11];
  const float* be2 = (const float*)d_in[12];
  const float* wpe = (const float*)d_in[13];
  const float* bpe = (const float*)d_in[14];
  const float* wq  = (const float*)d_in[15];
  const float* bq  = (const float*)d_in[16];
  const float* wk  = (const float*)d_in[17];
  const float* bk  = (const float*)d_in[18];
  const float* wv  = (const float*)d_in[19];
  const float* bv  = (const float*)d_in[20];
  const float* wo  = (const float*)d_in[21];
  const float* bo  = (const float*)d_in[22];
  const float* wh  = (const float*)d_in[23];
  const float* bh  = (const float*)d_in[24];
  float* out = (float*)d_out;

  char* ws = (char*)d_ws;
  int*      n_active = (int*)ws;        ws += 256;
  int*      idx      = (int*)ws;        ws += (size_t)NB * MM * 4;
  float*    feats    = (float*)ws;      ws += (size_t)NB * MM * CH * 4;
  float*    pos      = (float*)ws;      ws += (size_t)NB * MM * 2 * 4;
  _Float16* h16      = (_Float16*)ws;   ws += (size_t)NB * MM * CH * 2;
  _Float16* q16      = (_Float16*)ws;   ws += (size_t)NB * MM * ACH * 2;
  _Float16* k16      = (_Float16*)ws;   ws += (size_t)NB * MM * ACH * 2;
  _Float16* vT       = (_Float16*)ws;   ws += (size_t)NB * MM * ACH * 2;
  _Float16* o16      = (_Float16*)ws;   ws += (size_t)NB * MM * ACH * 2;
  _Float16* wqT      = (_Float16*)ws;   ws += 2 * CH * ACH * 2;
  _Float16* wkT      = (_Float16*)ws;   ws += 2 * CH * ACH * 2;
  _Float16* wvT      = (_Float16*)ws;   ws += 2 * CH * ACH * 2;
  _Float16* woT      = (_Float16*)ws;   ws += 2 * ACH * CH * 2;

  compact_kernel<<<NB, 256, 0, stream>>>(x, idx, n_active);
  feats_kernel<<<NB * MM, 64, 0, stream>>>(x, w0, b0, g0, be0, w1, b1, g1, be1,
                                           w2, b2, g2, be2, idx, n_active, feats, pos);
  wconv_kernel<<<256, 256, 0, stream>>>(wq, wk, wv, wo, wqT, wkT, wvT, woT);

  for (int layer = 0; layer < 2; ++layer) {
    h_kernel<<<(NB * MM * CH) / 256, 256, 0, stream>>>(feats, pos, n_active, wpe, bpe, layer, h16);
    qkv_gemm<<<(NB * 3 * 128 * 8) / 4, 128, 0, stream>>>(h16, wqT, wkT, wvT,
                                                         bq, bk, bv, layer, q16, k16, vT);
    flash_kernel<<<NB * 4 * 128, 32, 0, stream>>>(q16, k16, vT, n_active, o16);
    oproj_gemm<<<(NB * 128 * 4) / 4, 128, 0, stream>>>(o16, woT, bo, n_active, layer, feats);
  }

  pool_head<<<NB, 64, 0, stream>>>(feats, n_active, wh, bh, out);
}
